// SNNLanguageModel_61177514164307
// MI455X (gfx1250) — matrix operand order repaired and checked by reading, not compiled, not run on hardware
//
#include <hip/hip_runtime.h>

// ---------------------------------------------------------------------------
// Types for CDNA5 WMMA (wave32): V_WMMA_F32_16X16X32_BF16
// ---------------------------------------------------------------------------
typedef __attribute__((ext_vector_type(16))) __bf16 bf16x16;
typedef __attribute__((ext_vector_type(8)))  __bf16 bf16x8;
typedef __attribute__((ext_vector_type(8)))  float  f32x8;
typedef int v4i __attribute__((vector_size(16)));

#if defined(__gfx1250__) && __has_builtin(__builtin_amdgcn_global_load_async_to_lds_b128)
#define USE_ASYNC_LDS 1
#else
#define USE_ASYNC_LDS 0
#endif

__device__ __forceinline__ unsigned short f32_to_bf16(float f) {
  union { float f; unsigned int u; } x;
  x.f = f;
  unsigned int r = x.u + 0x7FFFu + ((x.u >> 16) & 1u);  // round-to-nearest-even
  return (unsigned short)(r >> 16);
}

// 16-byte global->LDS copy: async direct-to-LDS on gfx1250 if available,
// otherwise VGPR round-trip (global_load_b128 + ds_store_b128).
__device__ __forceinline__ void copy16_g2l(const unsigned short* g,
                                           unsigned short* l) {
#if USE_ASYNC_LDS
  auto gp = (__attribute__((address_space(1))) v4i*)(unsigned long long)g;
  auto lp = (__attribute__((address_space(3))) v4i*)(unsigned)(unsigned long long)l;
  __builtin_amdgcn_global_load_async_to_lds_b128(gp, lp, 0, 0);
#else
  *(uint4*)l = *(const uint4*)g;
#endif
}

__device__ __forceinline__ void wait_g2l() {
#if USE_ASYNC_LDS
#if __has_builtin(__builtin_amdgcn_s_wait_asynccnt)
  __builtin_amdgcn_s_wait_asynccnt(0);
#else
  asm volatile("s_wait_asynccnt 0" ::: "memory");
#endif
#endif
}

// ---------------------------------------------------------------------------
// f32 -> bf16 conversion, 4-wide
// ---------------------------------------------------------------------------
__global__ void cvt_f32_bf16_v4(const float4* __restrict__ src,
                                ushort4* __restrict__ dst, int n4) {
  int i = blockIdx.x * blockDim.x + threadIdx.x;
  if (i < n4) {
    float4 f = src[i];
    ushort4 o;
    o.x = f32_to_bf16(f.x);
    o.y = f32_to_bf16(f.y);
    o.z = f32_to_bf16(f.z);
    o.w = f32_to_bf16(f.w);
    dst[i] = o;
  }
}

// ---------------------------------------------------------------------------
// bf16 WMMA GEMM:  C[M,N] = A[M,K] * B[N,K]^T + bias[N]   (optional ReLU)
// Block: 256 threads = 8 waves (wave32). Block tile 128x128, K step 32,
// double-buffered LDS (one barrier per K step). Wave grid 2x4; each wave
// computes a 64x32 sub-tile = 4x2 WMMA 16x16 f32 accumulators (8 WMMA/step).
// ---------------------------------------------------------------------------
#define BM 128
#define BN 128
#define BK 32
#define LDT 40  // bf16 elems per LDS tile row (80B: 16B-aligned, bank-spread)

template <bool RELU>
__global__ void __launch_bounds__(256)
gemm_bf16_wmma(const unsigned short* __restrict__ A,   // [M,K] bf16 row-major
               const unsigned short* __restrict__ Bw,  // [N,K] bf16 row-major
               const float* __restrict__ bias,         // [N]
               float* __restrict__ C,                  // [M,N] f32
               int M, int N, int K) {
  __shared__ unsigned short As[2][BM * LDT];
  __shared__ unsigned short Bs[2][BN * LDT];

  const int tid   = threadIdx.x;
  const int wave  = tid >> 5;
  const int lane  = tid & 31;
  const int waveM = wave >> 2;  // 0..1 -> 64-row strips
  const int waveN = wave & 3;   // 0..3 -> 32-col strips
  const int lm    = lane & 15;  // row/col within 16
  const int lh    = lane >> 4;  // lane half selects K sub-chunk

  const int mBase = blockIdx.y * BM;
  const int nBase = blockIdx.x * BN;

  // Per-thread staging chunks: tiles are 128 rows x 32 cols = 512 x 16B.
  const int r0 = tid >> 2, c0 = tid & 3;              // chunk e = tid
  const int r1 = (tid + 256) >> 2, c1 = tid & 3;      // chunk e = tid + 256

  f32x8 zero = {};
  f32x8 acc[4][2];
#pragma unroll
  for (int i = 0; i < 4; ++i) { acc[i][0] = zero; acc[i][1] = zero; }

  // ---- stage first K slab into buffer 0 ----------------------------------
  copy16_g2l(A  + (size_t)(mBase + r0) * K + c0 * 8, &As[0][r0 * LDT + c0 * 8]);
  copy16_g2l(A  + (size_t)(mBase + r1) * K + c1 * 8, &As[0][r1 * LDT + c1 * 8]);
  copy16_g2l(Bw + (size_t)(nBase + r0) * K + c0 * 8, &Bs[0][r0 * LDT + c0 * 8]);
  copy16_g2l(Bw + (size_t)(nBase + r1) * K + c1 * 8, &Bs[0][r1 * LDT + c1 * 8]);
  wait_g2l();
  __syncthreads();

  const int nIter = K / BK;
  for (int it = 0; it < nIter; ++it) {
    const int cur = it & 1;
    const int kn  = (it + 1) * BK;

    // ---- stage next K slab into the other buffer (overlaps compute) ------
    if (kn < K) {
      const int nxt = cur ^ 1;
      copy16_g2l(A  + (size_t)(mBase + r0) * K + kn + c0 * 8,
                 &As[nxt][r0 * LDT + c0 * 8]);
      copy16_g2l(A  + (size_t)(mBase + r1) * K + kn + c1 * 8,
                 &As[nxt][r1 * LDT + c1 * 8]);
      copy16_g2l(Bw + (size_t)(nBase + r0) * K + kn + c0 * 8,
                 &Bs[nxt][r0 * LDT + c0 * 8]);
      copy16_g2l(Bw + (size_t)(nBase + r1) * K + kn + c1 * 8,
                 &Bs[nxt][r1 * LDT + c1 * 8]);
      if (kn + BK < K)  // speculative prefetch two slabs ahead
        __builtin_prefetch(A + (size_t)(mBase + r0) * K + kn + BK, 0, 1);
    }

    // ---- fragments per ISA 16-bit layouts --------------------------------
    // A 16x32: lane m=lane%16; VGPR0-3 K=(lh?8..15:0..7), VGPR4-7 K=+16.
    bf16x16 af[4];
#pragma unroll
    for (int sm = 0; sm < 4; ++sm) {
      int row = waveM * 64 + sm * 16 + lm;
      union { bf16x16 v; bf16x8 h[2]; } u;
      u.h[0] = *(const bf16x8*)(&As[cur][row * LDT + lh * 8]);
      u.h[1] = *(const bf16x8*)(&As[cur][row * LDT + 16 + lh * 8]);
      af[sm] = u.v;
    }
    // B 32x16: lane n=lane%16; VGPR0-3 K=(lh?8..15:0..7), VGPR4-7 K=+16.
    bf16x16 bfr[2];
#pragma unroll
    for (int sn = 0; sn < 2; ++sn) {
      int row = waveN * 32 + sn * 16 + lm;
      union { bf16x16 v; bf16x8 h[2]; } u;
      u.h[0] = *(const bf16x8*)(&Bs[cur][row * LDT + lh * 8]);
      u.h[1] = *(const bf16x8*)(&Bs[cur][row * LDT + 16 + lh * 8]);
      bfr[sn] = u.v;
    }

#pragma unroll
    for (int sm = 0; sm < 4; ++sm)
#pragma unroll
      for (int sn = 0; sn < 2; ++sn)
        acc[sm][sn] = __builtin_amdgcn_wmma_f32_16x16x32_bf16(
            false, af[sm], false, bfr[sn], (short)0, acc[sm][sn], false, false);

    wait_g2l();      // next-slab async copies landed in LDS
    __syncthreads(); // all waves done reading `cur`, next slab visible
  }

  // ---- epilogue: C/D layout lanes 0-15 -> M=r, lanes 16-31 -> M=r+8 ------
#pragma unroll
  for (int sm = 0; sm < 4; ++sm) {
#pragma unroll
    for (int sn = 0; sn < 2; ++sn) {
      int n = nBase + waveN * 32 + sn * 16 + lm;
      float bv = bias[n];
#pragma unroll
      for (int r = 0; r < 8; ++r) {
        int m = mBase + waveM * 64 + sm * 16 + lh * 8 + r;
        float v = acc[sm][sn][r] + bv;
        if (RELU) v = v > 0.f ? v : 0.f;
        C[(size_t)m * N + n] = v;
      }
    }
  }
}

// ---------------------------------------------------------------------------
// Leaky scan over time: v[t] = beta[h]*v[t-1] + y[t]; emit bf16 for next GEMM
// ---------------------------------------------------------------------------
__global__ void leaky_scan_to_bf16(const float* __restrict__ Y,
                                   const float* __restrict__ beta,
                                   unsigned short* __restrict__ Xb,
                                   int T, int BH, int Hmask) {
  int idx = blockIdx.x * blockDim.x + threadIdx.x;
  if (idx >= BH) return;
  float b = beta[idx & Hmask];  // H is a power of two
  float v = 0.f;
  size_t p = idx;
  for (int t = 0; t < T; ++t, p += (size_t)BH) {
    v = fmaf(b, v, Y[p]);
    Xb[p] = f32_to_bf16(v);
  }
}

// ---------------------------------------------------------------------------
// Host-side orchestration
// ---------------------------------------------------------------------------
extern "C" void kernel_launch(void* const* d_in, const int* in_sizes, int n_in,
                              void* d_out, int out_size, void* d_ws,
                              size_t ws_size, hipStream_t stream) {
  (void)in_sizes; (void)n_in; (void)out_size; (void)ws_size;

  const int T = 127, Bn = 256, V = 512, H = 1024;
  const int M = T * Bn;  // 32512, divisible by 128

  const float* x     = (const float*)d_in[0];
  const float* W1    = (const float*)d_in[1];
  const float* b1    = (const float*)d_in[2];
  const float* beta1 = (const float*)d_in[3];
  const float* W2    = (const float*)d_in[4];
  const float* b2    = (const float*)d_in[5];
  const float* beta2 = (const float*)d_in[6];
  const float* W3    = (const float*)d_in[7];
  const float* b3    = (const float*)d_in[8];
  const float* beta3 = (const float*)d_in[9];
  const float* W4    = (const float*)d_in[10];
  const float* b4    = (const float*)d_in[11];
  float* out = (float*)d_out;

  char* ws = (char*)d_ws;
  size_t off = 0;
  unsigned short* Xb  = (unsigned short*)(ws + off); off += (size_t)M * H * 2;
  float*          Y   = (float*)(ws + off);          off += (size_t)M * H * 4;
  unsigned short* W1b = (unsigned short*)(ws + off); off += (size_t)H * V * 2;
  unsigned short* W2b = (unsigned short*)(ws + off); off += (size_t)H * H * 2;
  unsigned short* W3b = (unsigned short*)(ws + off); off += (size_t)H * H * 2;
  unsigned short* W4b = (unsigned short*)(ws + off); off += (size_t)V * H * 2;

  auto cvt = [&](const float* s, unsigned short* d, int n) {
    int n4 = n / 4;  // all sizes are multiples of 4
    cvt_f32_bf16_v4<<<(n4 + 255) / 256, 256, 0, stream>>>(
        (const float4*)s, (ushort4*)d, n4);
  };
  cvt(x,  Xb,  M * V);
  cvt(W1, W1b, H * V);
  cvt(W2, W2b, H * H);
  cvt(W3, W3b, H * H);
  cvt(W4, W4b, V * H);

  const int BH = Bn * H;
  dim3 blk(256);

  // fc1 -> scan(beta1)
  gemm_bf16_wmma<false><<<dim3(H / BN, M / BM), blk, 0, stream>>>(
      Xb, W1b, b1, Y, M, H, V);
  leaky_scan_to_bf16<<<BH / 256, 256, 0, stream>>>(Y, beta1, Xb, T, BH, H - 1);

  // fc2 + ReLU -> scan(beta2)
  gemm_bf16_wmma<true><<<dim3(H / BN, M / BM), blk, 0, stream>>>(
      Xb, W2b, b2, Y, M, H, H);
  leaky_scan_to_bf16<<<BH / 256, 256, 0, stream>>>(Y, beta2, Xb, T, BH, H - 1);

  // fc3 + ReLU -> scan(beta3)
  gemm_bf16_wmma<true><<<dim3(H / BN, M / BM), blk, 0, stream>>>(
      Xb, W3b, b3, Y, M, H, H);
  leaky_scan_to_bf16<<<BH / 256, 256, 0, stream>>>(Y, beta3, Xb, T, BH, H - 1);

  // fc4 -> f32 output
  gemm_bf16_wmma<false><<<dim3(V / BN, M / BM), blk, 0, stream>>>(
      Xb, W4b, b4, out, M, V, H);
}
